// AttentionLayer_80161269613255
// MI455X (gfx1250) — matrix operand order repaired and checked by reading, not compiled, not run on hardware
//
#include <hip/hip_runtime.h>
#include <hip/hip_bf16.h>
#include <cstdint>

#define B_  4
#define C_  2048
#define D_  1024
#define H_  16
#define HD_ 64

typedef __attribute__((ext_vector_type(16))) __bf16 v16bf;
typedef __attribute__((ext_vector_type(8)))  __bf16 v8bf;
typedef __attribute__((ext_vector_type(8)))  float  v8f;

static __device__ inline v16bf mk16(v8bf lo, v8bf hi) {
  v16bf r;
#pragma unroll
  for (int i = 0; i < 8; ++i) { r[i] = lo[i]; r[i + 8] = hi[i]; }
  return r;
}

// CDNA5 async global->LDS copy (ASYNCcnt-tracked), GV addressing mode.
// lds_off = byte offset in LDS (low 32 bits of generic shared pointer).
static __device__ __forceinline__ void async_copy_b128(uint32_t lds_off,
                                                       const void* gaddr) {
  asm volatile("global_load_async_to_lds_b128 %0, %1, off"
               :: "v"(lds_off), "v"(gaddr)
               : "memory");
}
static __device__ __forceinline__ void wait_async0() {
  asm volatile("s_wait_asynccnt 0x0" ::: "memory");
}

// ---------------- elementwise helpers ----------------

__global__ void f32_to_bf16_kernel(const float* __restrict__ src,
                                   __bf16* __restrict__ dst, int n) {
  int i = blockIdx.x * blockDim.x + threadIdx.x;
  int stride = gridDim.x * blockDim.x;
  for (; i < n; i += stride) dst[i] = (__bf16)src[i];
}

__global__ void zero_f32_kernel(float* __restrict__ p, int n) {
  int i = blockIdx.x * blockDim.x + threadIdx.x;
  int stride = gridDim.x * blockDim.x;
  for (; i < n; i += stride) p[i] = 0.0f;
}

__global__ void scatter_keep_kernel(const long long* __restrict__ mi,
                                    float* __restrict__ keep) {
  int i = blockIdx.x * blockDim.x + threadIdx.x;
  if (i >= B_ * C_) return;
  long long v = mi[i];
  if (v >= 0) keep[(i / C_) * C_ + (int)v] = 1.0f;
}

// ---------------- generic bf16 WMMA GEMM (async double-buffered) ----------------
// C[M,N](f32 or bf16) = A[M,K](bf16) @ B[K,N](bf16) + bias[N](f32)
// block: 256 thr (8 waves), tile 128x128, BK=32, wave tile 32x64.
// A tile: async global->LDS copy; B tile: staged transposed via VGPRs.

template <bool OUT_F32>
__global__ __launch_bounds__(256) void gemm_bf16(
    const __bf16* __restrict__ A, const __bf16* __restrict__ Bm,
    const float* __restrict__ bias, void* __restrict__ Cout,
    int M, int N, int K) {
  __shared__ __align__(16) __bf16 As[2][128][32];   // [buf][m][k]
  __shared__ __align__(16) __bf16 Bt[2][128][32];   // [buf][n][k] (transposed)
  const int tid  = threadIdx.x;
  const int wave = tid >> 5, lane = tid & 31;
  const int hlf  = lane >> 4, l16 = lane & 15;
  const int waveM = wave >> 1, waveN = wave & 1;       // 4 x 2 wave grid
  const int blockM = blockIdx.y * 128, blockN = blockIdx.x * 128;

  v8f acc[2][4];
#pragma unroll
  for (int i = 0; i < 2; ++i)
#pragma unroll
    for (int j = 0; j < 4; ++j)
#pragma unroll
      for (int e = 0; e < 8; ++e) acc[i][j][e] = 0.0f;

  const int ar = tid >> 1, ac = (tid & 1) * 16;        // A tile: 2 thr/row, 16 elems
  const int bk = tid >> 3, bc = (tid & 7) * 16;        // B tile: 8 thr/row, 16 elems

  const __bf16* aRow = A + (size_t)(blockM + ar) * K + ac;
  const uint32_t asOff[2] = {
      (uint32_t)(uintptr_t)&As[0][ar][ac],
      (uint32_t)(uintptr_t)&As[1][ar][ac]};

  // ---- prologue: stage tile 0 ----
  async_copy_b128(asOff[0],      aRow);
  async_copy_b128(asOff[0] + 16, aRow + 8);
  v8bf b0, b1;
  {
    const v8bf* bp = (const v8bf*)(Bm + (size_t)bk * N + blockN + bc);
    b0 = bp[0]; b1 = bp[1];
  }
#pragma unroll
  for (int i = 0; i < 8; ++i) { Bt[0][bc + i][bk] = b0[i]; Bt[0][bc + 8 + i][bk] = b1[i]; }

  int buf = 0;
  for (int k0 = 0; k0 < K; k0 += 32, buf ^= 1) {
    wait_async0();        // own async writes into As[buf] complete
    __syncthreads();      // everyone's As[buf] + Bt[buf] published
    const int nxt = buf ^ 1;
    const bool more = (k0 + 32) < K;
    if (more) {           // prefetch next tile while computing this one
      async_copy_b128(asOff[nxt],      aRow + k0 + 32);
      async_copy_b128(asOff[nxt] + 16, aRow + k0 + 40);
      const v8bf* bp = (const v8bf*)(Bm + (size_t)(k0 + 32 + bk) * N + blockN + bc);
      b0 = bp[0]; b1 = bp[1];
    }

    v16bf af[2];
#pragma unroll
    for (int mt = 0; mt < 2; ++mt) {
      int row = waveM * 32 + mt * 16 + l16;
      // A 16x32 bf16 layout: lane<16 holds K 0-7 & 16-23; lane>=16 holds K 8-15 & 24-31
      v8bf lo = *(const v8bf*)&As[buf][row][hlf * 8];
      v8bf hi = *(const v8bf*)&As[buf][row][hlf * 8 + 16];
      af[mt] = mk16(lo, hi);
    }
    v16bf bfrag[4];
#pragma unroll
    for (int nt = 0; nt < 4; ++nt) {
      int col = waveN * 64 + nt * 16 + l16;
      // B 32x16 bf16 layout: lane<16 holds K 0-7 & 16-23; lane>=16 holds K 8-15 & 24-31 of its column
      v8bf lo = *(const v8bf*)&Bt[buf][col][hlf * 8];
      v8bf hi = *(const v8bf*)&Bt[buf][col][hlf * 8 + 16];
      bfrag[nt] = mk16(lo, hi);
    }
#pragma unroll
    for (int mt = 0; mt < 2; ++mt)
#pragma unroll
      for (int nt = 0; nt < 4; ++nt)
        acc[mt][nt] = __builtin_amdgcn_wmma_f32_16x16x32_bf16(
            false, af[mt], false, bfrag[nt], (short)0, acc[mt][nt], false, false);

    if (more) {  // buffer nxt is free: all waves finished reading it before the barrier
#pragma unroll
      for (int i = 0; i < 8; ++i) { Bt[nxt][bc + i][bk] = b0[i]; Bt[nxt][bc + 8 + i][bk] = b1[i]; }
    }
  }

  // epilogue: C frag VGPR r -> row = base + r + hlf*8, col = base + l16
#pragma unroll
  for (int mt = 0; mt < 2; ++mt) {
    int row0 = blockM + waveM * 32 + mt * 16 + hlf * 8;
#pragma unroll
    for (int nt = 0; nt < 4; ++nt) {
      int col = blockN + waveN * 64 + nt * 16 + l16;
      float bv = bias[col];
#pragma unroll
      for (int r = 0; r < 8; ++r) {
        float v = acc[mt][nt][r] + bv;
        size_t idx = (size_t)(row0 + r) * N + col;
        if (OUT_F32) ((float*)Cout)[idx] = v;
        else         ((__bf16*)Cout)[idx] = (__bf16)v;
      }
    }
  }
}

// ---------------- flash attention (bf16 WMMA, fp32 softmax) ----------------
// grid: (C/128, B*H). 8 waves x 16 query rows. 64-key blocks.

__global__ __launch_bounds__(256) void flash_attn_kernel(
    const __bf16* __restrict__ qkv, const float* __restrict__ keep,
    __bf16* __restrict__ O) {
  __shared__ __align__(16) __bf16 Vt[64][64];      // [hd][key]
  __shared__ __align__(16) __bf16 Ps[8][16][64];   // per-wave P strip
  const int tid  = threadIdx.x;
  const int wave = tid >> 5, lane = tid & 31;
  const int hlf  = lane >> 4, l16 = lane & 15;
  const int bh = blockIdx.y, b = bh / H_, h = bh % H_;
  const size_t rowStride = 3 * D_;
  const __bf16* Qb = qkv + (size_t)b * C_ * rowStride + h * HD_;
  const __bf16* Kb = Qb + D_;
  const __bf16* Vb = Qb + 2 * D_;
  const int q0 = blockIdx.x * 128 + wave * 16;
  const float* keepB = keep + b * C_;

  // Q fragments (16 rows x 64, K split 2x32), loaded once
  v16bf qf[2];
  {
    const __bf16* qp = Qb + (size_t)(q0 + l16) * rowStride;
#pragma unroll
    for (int kk = 0; kk < 2; ++kk) {
      v8bf lo = *(const v8bf*)(qp + kk * 32 + hlf * 8);
      v8bf hi = *(const v8bf*)(qp + kk * 32 + hlf * 8 + 16);
      qf[kk] = mk16(lo, hi);
    }
  }

  float m_i[8], l_i[8];
  v8f oacc[4];
#pragma unroll
  for (int r = 0; r < 8; ++r) { m_i[r] = -1e30f; l_i[r] = 0.0f; }
#pragma unroll
  for (int ct = 0; ct < 4; ++ct)
#pragma unroll
    for (int r = 0; r < 8; ++r) oacc[ct][r] = 0.0f;

  const int vr = tid >> 2, vc = (tid & 3) * 16;  // V block staging: 4 thr/row

  for (int kb = 0; kb < C_; kb += 64) {
    __syncthreads();
    {  // stage V[64 keys][64 hd] transposed into LDS
      const v8bf* vp = (const v8bf*)(Vb + (size_t)(kb + vr) * rowStride + vc);
      v8bf u0 = vp[0], u1 = vp[1];
#pragma unroll
      for (int i = 0; i < 8; ++i) { Vt[vc + i][vr] = u0[i]; Vt[vc + 8 + i][vr] = u1[i]; }
    }
    __syncthreads();

    // S = Q @ K^T (K rows read directly from global: contiguous per lane)
    v8f s[4];
#pragma unroll
    for (int ct = 0; ct < 4; ++ct) {
      v8f accv;
#pragma unroll
      for (int e = 0; e < 8; ++e) accv[e] = 0.0f;
      const __bf16* kp = Kb + (size_t)(kb + ct * 16 + l16) * rowStride;
#pragma unroll
      for (int kk = 0; kk < 2; ++kk) {
        v8bf lo = *(const v8bf*)(kp + kk * 32 + hlf * 8);
        v8bf hi = *(const v8bf*)(kp + kk * 32 + hlf * 8 + 16);
        v16bf kf = mk16(lo, hi);
        accv = __builtin_amdgcn_wmma_f32_16x16x32_bf16(
            false, qf[kk], false, kf, (short)0, accv, false, false);
      }
      float cmask = (keepB[kb + ct * 16 + l16] > 0.0f) ? 0.0f : -1e30f;
#pragma unroll
      for (int e = 0; e < 8; ++e) s[ct][e] = accv[e] * 0.125f + cmask;
    }

    // online softmax: row (r + hlf*8) values live across the 16 lanes of a half
    float corr[8];
#pragma unroll
    for (int r = 0; r < 8; ++r) {
      float bm = fmaxf(fmaxf(s[0][r], s[1][r]), fmaxf(s[2][r], s[3][r]));
#pragma unroll
      for (int d = 1; d < 16; d <<= 1) bm = fmaxf(bm, __shfl_xor(bm, d, 32));
      float mnew = fmaxf(m_i[r], bm);
      corr[r] = __expf(m_i[r] - mnew);
      m_i[r] = mnew;
    }
#pragma unroll
    for (int r = 0; r < 8; ++r) {
      float rs = 0.0f;
#pragma unroll
      for (int ct = 0; ct < 4; ++ct) {
        float p = __expf(s[ct][r] - m_i[r]);
        s[ct][r] = p;
        rs += p;
      }
#pragma unroll
      for (int d = 1; d < 16; d <<= 1) rs += __shfl_xor(rs, d, 32);
      l_i[r] = l_i[r] * corr[r] + rs;
    }
#pragma unroll
    for (int ct = 0; ct < 4; ++ct)
#pragma unroll
      for (int r = 0; r < 8; ++r) oacc[ct][r] *= corr[r];

    // C-layout -> A-layout via per-wave LDS strip (same-wave, no barrier needed)
#pragma unroll
    for (int ct = 0; ct < 4; ++ct)
#pragma unroll
      for (int r = 0; r < 8; ++r)
        Ps[wave][r + hlf * 8][ct * 16 + l16] = (__bf16)s[ct][r];
    v16bf pf[2];
#pragma unroll
    for (int kk = 0; kk < 2; ++kk) {
      v8bf lo = *(const v8bf*)&Ps[wave][l16][kk * 32 + hlf * 8];
      v8bf hi = *(const v8bf*)&Ps[wave][l16][kk * 32 + hlf * 8 + 16];
      pf[kk] = mk16(lo, hi);
    }

    // O += P @ V
#pragma unroll
    for (int ct = 0; ct < 4; ++ct)
#pragma unroll
      for (int kk = 0; kk < 2; ++kk) {
        v8bf lo = *(const v8bf*)&Vt[ct * 16 + l16][kk * 32 + hlf * 8];
        v8bf hi = *(const v8bf*)&Vt[ct * 16 + l16][kk * 32 + hlf * 8 + 16];
        v16bf vf = mk16(lo, hi);
        oacc[ct] = __builtin_amdgcn_wmma_f32_16x16x32_bf16(
            false, pf[kk], false, vf, (short)0, oacc[ct], false, false);
      }
  }

  // finalize: 1/l and post-softmax row mask folded into store
#pragma unroll
  for (int r = 0; r < 8; ++r) {
    int row = q0 + r + hlf * 8;
    float sc = (keepB[row] > 0.0f ? 1.0f : 0.0f) / fmaxf(l_i[r], 1e-20f);
#pragma unroll
    for (int ct = 0; ct < 4; ++ct)
      O[((size_t)b * C_ + row) * D_ + h * HD_ + ct * 16 + l16] =
          (__bf16)(oacc[ct][r] * sc);
  }
}

// ---------------- launcher ----------------

extern "C" void kernel_launch(void* const* d_in, const int* in_sizes, int n_in,
                              void* d_out, int out_size, void* d_ws, size_t ws_size,
                              hipStream_t stream) {
  (void)in_sizes; (void)n_in; (void)out_size; (void)ws_size;
  const float*     x        = (const float*)d_in[0];
  const long long* mask_ind = (const long long*)d_in[1];
  const float*     Wqkv     = (const float*)d_in[2];
  const float*     bqkv     = (const float*)d_in[3];
  const float*     Wproj    = (const float*)d_in[4];
  const float*     bproj    = (const float*)d_in[5];
  float*           out      = (float*)d_out;

  char* ws = (char*)d_ws;
  size_t off = 0;
  auto alloc = [&](size_t bytes) -> void* {
    void* p = ws + off;
    off += (bytes + 255) & ~(size_t)255;
    return p;
  };
  __bf16* xb   = (__bf16*)alloc((size_t)B_ * C_ * D_ * 2);
  __bf16* wqb  = (__bf16*)alloc((size_t)D_ * 3 * D_ * 2);
  __bf16* wpb  = (__bf16*)alloc((size_t)D_ * D_ * 2);
  __bf16* qkvb = (__bf16*)alloc((size_t)B_ * C_ * 3 * D_ * 2);
  __bf16* ob   = (__bf16*)alloc((size_t)B_ * C_ * D_ * 2);
  float*  keep = (float*)alloc((size_t)B_ * C_ * 4);

  f32_to_bf16_kernel<<<1024, 256, 0, stream>>>(x, xb, B_ * C_ * D_);
  f32_to_bf16_kernel<<<512, 256, 0, stream>>>(Wqkv, wqb, D_ * 3 * D_);
  f32_to_bf16_kernel<<<256, 256, 0, stream>>>(Wproj, wpb, D_ * D_);
  zero_f32_kernel<<<32, 256, 0, stream>>>(keep, B_ * C_);
  scatter_keep_kernel<<<(B_ * C_ + 255) / 256, 256, 0, stream>>>(mask_ind, keep);

  // qkv = x @ Wqkv + bqkv : M=8192, N=3072, K=1024 (bf16 out)
  gemm_bf16<false><<<dim3(3 * D_ / 128, B_ * C_ / 128), 256, 0, stream>>>(
      xb, wqb, bqkv, qkvb, B_ * C_, 3 * D_, D_);

  // fused masked attention -> O (bf16, [B,C,D])
  flash_attn_kernel<<<dim3(C_ / 128, B_ * H_), 256, 0, stream>>>(qkvb, keep, ob);

  // out = O @ Wproj + bproj : M=8192, N=1024, K=1024 (f32 out)
  gemm_bf16<true><<<dim3(D_ / 128, B_ * C_ / 128), 256, 0, stream>>>(
      ob, wpb, bproj, out, B_ * C_, D_, D_);
}
